// LoRAModulatedAllegro_10720238371314
// MI455X (gfx1250) — compile-verified, matrix operand order checked by reading, not run on hardware
//
#include <hip/hip_runtime.h>
#include <hip/hip_bf16.h>
#include <math.h>

#define E_ 65536
#define N_ 2048
#define H_ 256
#define C_ 32
#define P_ 64
#define R_ 4
#define OUT_ 128
#define EPS_ 1e-3f
#define SQRT3_ 1.7320508075688772f
#define INV3_ 0.5773502691896258f

typedef __bf16 bf16_t;
typedef __attribute__((ext_vector_type(16))) __bf16 v16bf;
typedef __attribute__((ext_vector_type(8)))  __bf16 v8bf;
typedef __attribute__((ext_vector_type(8)))  float  v8f;

__device__ __forceinline__ bf16_t f2bf(float f) {
  unsigned u = __builtin_bit_cast(unsigned, f);
  unsigned r = u + 0x7FFFu + ((u >> 16) & 1u);
  unsigned short h = (unsigned short)(r >> 16);
  return __builtin_bit_cast(bf16_t, h);
}
__device__ __forceinline__ float siluf(float x) { return x / (1.0f + expf(-x)); }

// ---------------------------------------------------------------------------
// Edge-parallel GEMM: D[E x M] = epilogue(A[E x K](bf16) @ W[K x M](f32)).
// Block = 256 threads (8 waves); block tile = 256 rows x (CT*16) cols.
// Each wave owns 2 row tiles x CT col tiles (2*CT accumulators):
//   per 32-k step: 4 global b128 (A) + 2*CT ds b128 (B) -> 2*CT WMMAs,
//   i.e. 1 ds_b128 per WMMA and each B fragment feeds 2 WMMAs.
// W slab (K x CT*16) converted f32->bf16 into LDS, N-major for b128 ds reads.
// epilogue: v = acc; v += add1; v *= rowscale*alpha; v += add2; v *= beta; act?
// ---------------------------------------------------------------------------
template <int CT>
__global__ __launch_bounds__(256) void k_gemm(
    const bf16_t* __restrict__ A, int lda,
    const float* __restrict__ W, int K, int M,
    const float* __restrict__ add1,
    const float* __restrict__ add2,
    const float* __restrict__ rowscale,
    float alpha, float beta, int act,
    float* __restrict__ outF, bf16_t* __restrict__ outB, int ldB)
{
  __shared__ bf16_t Wl[(320 + 8) * (CT * 16)];
  const int Kp = K + 8;
  const int tid = threadIdx.x;
  const int col0 = blockIdx.y * (CT * 16);
  const int row0 = blockIdx.x * 256;

  for (int i = tid; i < K * (CT * 16); i += 256) {
    int c = i / K, k = i - c * K;
    Wl[c * Kp + k] = f2bf(W[(size_t)k * M + col0 + c]);
  }
  __syncthreads();

  const int lane = tid & 31;
  const int wid  = tid >> 5;
  const int g    = lane >> 4;
  const int n    = lane & 15;
  const bf16_t* ap0 = A + (size_t)(row0 + wid * 32 + n) * lda;
  const bf16_t* ap1 = A + (size_t)(row0 + wid * 32 + 16 + n) * lda;

  v8f acc[2][CT];
#pragma unroll
  for (int rt = 0; rt < 2; ++rt)
#pragma unroll
    for (int t = 0; t < CT; ++t) acc[rt][t] = (v8f){};

  for (int kk = 0; kk < K; kk += 32) {
    v8bf a0lo = *(const v8bf*)(ap0 + kk + 8 * g);
    v8bf a0hi = *(const v8bf*)(ap0 + kk + 16 + 8 * g);
    v8bf a1lo = *(const v8bf*)(ap1 + kk + 8 * g);
    v8bf a1hi = *(const v8bf*)(ap1 + kk + 16 + 8 * g);
    v16bf av0, av1;
#pragma unroll
    for (int i = 0; i < 8; ++i) {
      av0[i] = a0lo[i]; av0[8 + i] = a0hi[i];
      av1[i] = a1lo[i]; av1[8 + i] = a1hi[i];
    }
#pragma unroll
    for (int t = 0; t < CT; ++t) {
      const bf16_t* bp = Wl + (t * 16 + n) * Kp;
      v8bf blo = *(const v8bf*)(bp + kk + 8 * g);
      v8bf bhi = *(const v8bf*)(bp + kk + 16 + 8 * g);
      v16bf bv;
#pragma unroll
      for (int i = 0; i < 8; ++i) { bv[i] = blo[i]; bv[8 + i] = bhi[i]; }
      acc[0][t] = __builtin_amdgcn_wmma_f32_16x16x32_bf16(false, av0, false, bv,
                                                          (short)0, acc[0][t],
                                                          false, false);
      acc[1][t] = __builtin_amdgcn_wmma_f32_16x16x32_bf16(false, av1, false, bv,
                                                          (short)0, acc[1][t],
                                                          false, false);
    }
  }

#pragma unroll
  for (int rt = 0; rt < 2; ++rt) {
#pragma unroll
    for (int t = 0; t < CT; ++t) {
#pragma unroll
      for (int r = 0; r < 8; ++r) {
        int row = row0 + wid * 32 + rt * 16 + g * 8 + r;
        int col = col0 + t * 16 + n;
        float v = acc[rt][t][r];
        if (add1) v += add1[(size_t)row * M + col];
        float rs = rowscale ? rowscale[row] : 1.0f;
        v = v * rs * alpha;
        if (add2) v += add2[(size_t)row * M + col];
        v *= beta;
        if (act) v = siluf(v);
        if (outF) outF[(size_t)row * M + col] = v;
        if (outB) outB[(size_t)row * ldB + col] = f2bf(v);
      }
    }
  }
}

// --------------------------- small helper kernels ---------------------------

__global__ void k_y1(const float* __restrict__ pos, float* __restrict__ Y1) {
  int e = blockIdx.x * blockDim.x + threadIdx.x;
  if (e >= E_) return;
  float px = pos[e * 3 + 0] + EPS_;
  float py = pos[e * 3 + 1] + EPS_;
  float pz = pos[e * 3 + 2] + EPS_;
  float inv = rsqrtf(px * px + py * py + pz * pz);
  Y1[e * 3 + 0] = SQRT3_ * px * inv;
  Y1[e * 3 + 1] = SQRT3_ * py * inv;
  Y1[e * 3 + 2] = SQRT3_ * pz * inv;
}

__global__ void k_nn(const int* __restrict__ idx, const float* __restrict__ mask,
                     float* __restrict__ segN) {
  int e = blockIdx.x * blockDim.x + threadIdx.x;
  if (e >= E_) return;
  atomicAdd(&segN[idx[e]], mask[e]);
}

// x0 = [ev*m, nv[s], nv[r]] * (pos_length != 0), bf16; one thread per (e, j<64)
__global__ void k_x0(const float* __restrict__ ev, const float* __restrict__ nv,
                     const float* __restrict__ pl, const float* __restrict__ em,
                     const float* __restrict__ nm, const int* __restrict__ idx,
                     bf16_t* __restrict__ X0) {
  int t = blockIdx.x * blockDim.x + threadIdx.x;
  int e = t >> 6, j = t & 63;
  if (e >= E_) return;
  float z = (pl[e] == 0.0f) ? 0.0f : 1.0f;
  int s = idx[e], r = idx[E_ + e];
  bf16_t* row = X0 + (size_t)e * 192;
  row[j]       = f2bf(ev[(size_t)e * 64 + j] * em[e] * z);
  row[64 + j]  = f2bf(nv[(size_t)s * 64 + j] * nm[s] * z);
  row[128 + j] = f2bf(nv[(size_t)r * 64 + j] * nm[r] * z);
}

// split w0 -> Vs, Vv; one thread per (e, c<32)
__global__ void k_w0split(const float* __restrict__ w0, const float* __restrict__ Y1,
                          float* __restrict__ Vs, float* __restrict__ Vv) {
  int t = blockIdx.x * blockDim.x + threadIdx.x;
  int e = t >> 5, c = t & 31;
  if (e >= E_) return;
  Vs[(size_t)e * 32 + c] = w0[(size_t)e * 64 + c];
  float wv = w0[(size_t)e * 64 + 32 + c];
  for (int d = 0; d < 3; ++d)
    Vv[((size_t)e * 32 + c) * 3 + d] = wv * Y1[e * 3 + d];
}

// xm(bf16) = x * mask into 320-wide arena cols [0,256); one thread per (e,c<256)
__global__ void k_xm(const float* __restrict__ x, const float* __restrict__ em,
                     bf16_t* __restrict__ arena) {
  int t = blockIdx.x * blockDim.x + threadIdx.x;
  int e = t >> 8, c = t & 255;
  if (e >= E_) return;
  arena[(size_t)e * 320 + c] = f2bf(x[(size_t)e * 256 + c] * em[e]);
}

// LoRA low-rank delta: delta = (x@A * silu(mod@G)) @ B  (R=4).
// One wave32 per edge: lanes split K (coalesced), butterfly-reduce partials.
// x = [x1*scale1 (len1, stride ld1) | x2 (len2, tight)]
__global__ __launch_bounds__(256) void k_delta(
    const float* __restrict__ x1, int ld1, int len1,
    const float* __restrict__ scale1,
    const float* __restrict__ x2, int len2,
    const float* __restrict__ mod, const float* __restrict__ G,
    const float* __restrict__ Aw, const float* __restrict__ B,
    int M, float* __restrict__ delta) {
  int lane = threadIdx.x & 31;
  int e = blockIdx.x * 8 + (threadIdx.x >> 5);
  if (e >= E_) return;
  float s[R_] = {0.f, 0.f, 0.f, 0.f};
  for (int p = lane; p < P_; p += 32) {
    float mv = mod[(size_t)e * P_ + p];
#pragma unroll
    for (int r = 0; r < R_; ++r) s[r] += mv * G[p * R_ + r];
  }
  float t[R_] = {0.f, 0.f, 0.f, 0.f};
  float sc = scale1 ? scale1[e] : 1.0f;
  const float* xr = x1 + (size_t)e * ld1;
  for (int k = lane; k < len1; k += 32) {
    float xv = xr[k] * sc;
#pragma unroll
    for (int r = 0; r < R_; ++r) t[r] += xv * Aw[k * R_ + r];
  }
  if (x2) {
    const float* x2r = x2 + (size_t)e * len2;
    for (int k = lane; k < len2; k += 32) {
      float xv = x2r[k];
#pragma unroll
      for (int r = 0; r < R_; ++r) t[r] += xv * Aw[(len1 + k) * R_ + r];
    }
  }
#pragma unroll
  for (int off = 16; off; off >>= 1) {
#pragma unroll
    for (int r = 0; r < R_; ++r) {
      s[r] += __shfl_xor(s[r], off, 32);
      t[r] += __shfl_xor(t[r], off, 32);
    }
  }
#pragma unroll
  for (int r = 0; r < R_; ++r) t[r] *= siluf(s[r]);
  float* dr = delta + (size_t)e * M;
  for (int j = lane; j < M; j += 32) {
    float d = 0.f;
#pragma unroll
    for (int r = 0; r < R_; ++r) d += t[r] * B[r * M + j];
    dr[j] = d;
  }
}

// scatter segment sums; one thread per (e, c<32)
__global__ void k_scatter(const float* __restrict__ w, const float* __restrict__ em,
                          const float* __restrict__ Y1, const int* __restrict__ idx,
                          float* __restrict__ segS, float* __restrict__ segV) {
  int t = blockIdx.x * blockDim.x + threadIdx.x;
  int e = t >> 5, c = t & 31;
  if (e >= E_) return;
  float m = em[e];
  int s = idx[e];
  atomicAdd(&segS[(size_t)s * 32 + c], w[(size_t)e * 64 + c] * m);
  float wvv = w[(size_t)e * 64 + 32 + c] * m;
  for (int d = 0; d < 3; ++d)
    atomicAdd(&segV[((size_t)s * 32 + c) * 3 + d], wvv * Y1[e * 3 + d]);
}

// gather + build out0; one thread per (e, c<32)
__global__ void k_out0(const int* __restrict__ idx, const float* __restrict__ segN,
                       const float* __restrict__ segS, const float* __restrict__ segV,
                       const float* __restrict__ Vs, const float* __restrict__ Vv,
                       float* __restrict__ wYs, float* __restrict__ wYv,
                       float* __restrict__ out0f, bf16_t* __restrict__ arena) {
  int t = blockIdx.x * blockDim.x + threadIdx.x;
  int e = t >> 5, c = t & 31;
  if (e >= E_) return;
  int s = idx[e];
  float denom = segN[s] + 1e-5f;
  float wys = segS[(size_t)s * 32 + c] / denom;
  wYs[(size_t)e * 32 + c] = wys;
  float accd = 0.f;
  for (int d = 0; d < 3; ++d) {
    float wyv = segV[((size_t)s * 32 + c) * 3 + d] / denom;
    wYv[((size_t)e * 32 + c) * 3 + d] = wyv;
    accd += wyv * Vv[((size_t)e * 32 + c) * 3 + d];
  }
  float lo = wys * Vs[(size_t)e * 32 + c];
  float hi = INV3_ * accd;
  out0f[(size_t)e * 64 + c]      = lo;
  out0f[(size_t)e * 64 + 32 + c] = hi;
  arena[(size_t)e * 320 + 256 + c] = f2bf(lo);
  arena[(size_t)e * 320 + 288 + c] = f2bf(hi);
}

// Vv' = einsum('ecd,cf->efd', [wYs*Vv ; Vs*wYv], lin_V); one thread per (e, f<32)
__global__ void k_vv(const float* __restrict__ wYs, const float* __restrict__ wYv,
                     const float* __restrict__ Vs, const float* __restrict__ Vv,
                     const float* __restrict__ linV, float* __restrict__ Vv2) {
  int t = blockIdx.x * blockDim.x + threadIdx.x;
  int e = t >> 5, f = t & 31;
  if (e >= E_) return;
  float acc[3] = {0.f, 0.f, 0.f};
  for (int c = 0; c < 32; ++c) {
    float lv = linV[c * 32 + f];
    float wys = wYs[(size_t)e * 32 + c];
    for (int d = 0; d < 3; ++d)
      acc[d] += wys * Vv[((size_t)e * 32 + c) * 3 + d] * lv;
  }
  for (int c = 0; c < 32; ++c) {
    float lv = linV[(32 + c) * 32 + f];
    float vs = Vs[(size_t)e * 32 + c];
    for (int d = 0; d < 3; ++d)
      acc[d] += vs * wYv[((size_t)e * 32 + c) * 3 + d] * lv;
  }
  for (int d = 0; d < 3; ++d)
    Vv2[((size_t)e * 32 + f) * 3 + d] = acc[d];
}

// W_comb = W_post(256x128) @ W_final[:128,:](128x128); one thread per (i,j)
__global__ void k_ww(const float* __restrict__ Wp, const float* __restrict__ Wf,
                     float* __restrict__ Wc) {
  int t = blockIdx.x * blockDim.x + threadIdx.x;
  int i = t >> 7, j = t & 127;
  if (i >= 256) return;
  float acc = 0.f;
  for (int k = 0; k < 128; ++k) acc += Wp[i * 128 + k] * Wf[k * 128 + j];
  Wc[i * 128 + j] = acc;
}

// ---------------------------------------------------------------------------

extern "C" void kernel_launch(void* const* d_in, const int* in_sizes, int n_in,
                              void* d_out, int out_size, void* d_ws, size_t ws_size,
                              hipStream_t stream) {
  const float* node_vec  = (const float*)d_in[0];
  const float* edge_vec  = (const float*)d_in[1];
  const float* pos_vec   = (const float*)d_in[2];
  const float* pos_len   = (const float*)d_in[3];
  const float* edge_cut  = (const float*)d_in[4];
  const float* edge_mask = (const float*)d_in[5];
  const int*   edge_idx  = (const int*)  d_in[6];
  const float* node_mask = (const float*)d_in[7];
  const float* mod       = (const float*)d_in[8];
  const float* We0       = (const float*)d_in[9];
  const float* We1       = (const float*)d_in[10];
  const float* We2       = (const float*)d_in[11];
  const float* We3       = (const float*)d_in[12];
  const float* Ww_init   = (const float*)d_in[13];
  const float* lw_W      = (const float*)d_in[14];
  const float* lw_A      = (const float*)d_in[15];
  const float* lw_G      = (const float*)d_in[16];
  const float* lw_B      = (const float*)d_in[17];
  const float* m2_W0     = (const float*)d_in[18];
  const float* m2_A0     = (const float*)d_in[19];
  const float* m2_G0     = (const float*)d_in[20];
  const float* m2_B0     = (const float*)d_in[21];
  const float* m2_W1     = (const float*)d_in[22];
  const float* m2_A1     = (const float*)d_in[23];
  const float* m2_G1     = (const float*)d_in[24];
  const float* m2_B1     = (const float*)d_in[25];
  const float* lin_V     = (const float*)d_in[26];
  const float* W_post    = (const float*)d_in[27];
  const float* W_final   = (const float*)d_in[28];
  float* out = (float*)d_out;

  // ---- workspace carve-up ----
  char* ws = (char*)d_ws;
  size_t off = 0;
  auto alloc = [&](size_t bytes) -> void* {
    void* p = (void*)(ws + off);
    off += (bytes + 255) & ~(size_t)255;
    return p;
  };
  float* F_X     = (float*)alloc((size_t)E_ * 256 * 4);
  float* F_H     = (float*)alloc((size_t)E_ * 256 * 4);
  float* F_DELTA = (float*)alloc((size_t)E_ * 256 * 4);
  float* F_W     = (float*)alloc((size_t)E_ * 64 * 4);
  float* F_OUT0  = (float*)alloc((size_t)E_ * 64 * 4);
  float* F_VS    = (float*)alloc((size_t)E_ * 32 * 4);
  float* F_VVA   = (float*)alloc((size_t)E_ * 96 * 4);
  float* F_VVB   = (float*)alloc((size_t)E_ * 96 * 4);
  float* F_WYS   = (float*)alloc((size_t)E_ * 32 * 4);
  float* F_WYV   = (float*)alloc((size_t)E_ * 96 * 4);
  float* F_Y1    = (float*)alloc((size_t)E_ * 3 * 4);
  float* F_WC    = (float*)alloc((size_t)256 * 128 * 4);
  float* SEG_N   = (float*)alloc((size_t)N_ * 4);
  float* SEG_S   = (float*)alloc((size_t)N_ * 32 * 4);
  float* SEG_V   = (float*)alloc((size_t)N_ * 96 * 4);
  bf16_t* BF_X0  = (bf16_t*)alloc((size_t)E_ * 192 * 2);
  bf16_t* BF_X1  = (bf16_t*)alloc((size_t)E_ * 32 * 2);
  bf16_t* BF_X2  = (bf16_t*)alloc((size_t)E_ * 64 * 2);
  bf16_t* BF_X3  = (bf16_t*)alloc((size_t)E_ * 128 * 2);
  bf16_t* BF_X   = (bf16_t*)alloc((size_t)E_ * 256 * 2);
  bf16_t* ARENA1 = (bf16_t*)alloc((size_t)E_ * 320 * 2);  // [xm | out0]
  bf16_t* ARENA2 = (bf16_t*)alloc((size_t)E_ * 256 * 2);  // h (bf16)
  (void)in_sizes; (void)n_in; (void)out_size; (void)ws_size;

  auto gemm = [&](const bf16_t* A, int lda, const float* W, int K, int M,
                  const float* add1, const float* add2, const float* rowscale,
                  float alpha, float beta, int act,
                  float* outF, bf16_t* outB, int ldB) {
    if (M >= 64) {
      dim3 grid(E_ / 256, M / 64);
      k_gemm<4><<<grid, dim3(256), 0, stream>>>(A, lda, W, K, M, add1, add2,
                                                rowscale, alpha, beta, act,
                                                outF, outB, ldB);
    } else {  // M == 32
      dim3 grid(E_ / 256, 1);
      k_gemm<2><<<grid, dim3(256), 0, stream>>>(A, lda, W, K, M, add1, add2,
                                                rowscale, alpha, beta, act,
                                                outF, outB, ldB);
    }
  };

  const int TPB = 256;
  const dim3 gE(E_ / TPB);           // per-edge threads
  const dim3 gEW(E_ / 8);            // per-edge waves (k_delta)
  const dim3 gE32(E_ * 32 / TPB);    // per (edge, c<32)
  const dim3 gE64(E_ * 64 / TPB);    // per (edge, j<64)
  const dim3 gE256(E_ * 256 / TPB);  // per (edge, c<256)

  // ---- prep ----
  hipMemsetAsync(SEG_N, 0, (size_t)N_ * 4, stream);
  k_nn<<<gE, TPB, 0, stream>>>(edge_idx, edge_mask, SEG_N);
  k_y1<<<gE, TPB, 0, stream>>>(pos_vec, F_Y1);
  k_x0<<<gE64, TPB, 0, stream>>>(edge_vec, node_vec, pos_len, edge_mask,
                                 node_mask, edge_idx, BF_X0);
  k_ww<<<dim3(256 * 128 / TPB), TPB, 0, stream>>>(W_post, W_final, F_WC);

  // ---- edge MLP ----
  gemm(BF_X0, 192, We0, 192, 32,  nullptr, nullptr, nullptr, 1.f, 1.f, 1, nullptr, BF_X1, 32);
  gemm(BF_X1, 32,  We1, 32,  64,  nullptr, nullptr, nullptr, 1.f, 1.f, 1, nullptr, BF_X2, 64);
  gemm(BF_X2, 64,  We2, 64,  128, nullptr, nullptr, nullptr, 1.f, 1.f, 1, nullptr, BF_X3, 128);
  gemm(BF_X3, 128, We3, 128, 256, nullptr, nullptr, edge_cut, 1.f, 1.f, 0, F_X, BF_X, 256);

  // ---- w0 -> Vs, Vv ----
  gemm(BF_X, 256, Ww_init, 256, 64, nullptr, nullptr, nullptr, 1.f, 1.f, 0, F_W, nullptr, 0);
  k_w0split<<<gE32, TPB, 0, stream>>>(F_W, F_Y1, F_VS, F_VVA);

  float* Vv  = F_VVA;
  float* Vv2 = F_VVB;
  const float invb = 0.8944271909999159f;  // 1/sqrt(1 + 0.5^2)

  for (int l = 0; l < 3; ++l) {
    // xm (bf16) into arena cols [0,256)
    k_xm<<<gE256, TPB, 0, stream>>>(F_X, edge_mask, ARENA1);

    // w = lora(xm, lw)
    k_delta<<<gEW, TPB, 0, stream>>>(F_X, 256, 256, edge_mask, nullptr, 0, mod,
                                     lw_G + l * 256, lw_A + l * 1024,
                                     lw_B + l * 256, 64, F_DELTA);
    gemm(ARENA1, 320, lw_W + l * 16384, 256, 64, F_DELTA, nullptr, nullptr,
         1.f, 1.f, 0, F_W, nullptr, 0);

    // segment sums over senders
    hipMemsetAsync(SEG_S, 0, (size_t)N_ * 32 * 4, stream);
    hipMemsetAsync(SEG_V, 0, (size_t)N_ * 96 * 4, stream);
    k_scatter<<<gE32, TPB, 0, stream>>>(F_W, edge_mask, F_Y1, edge_idx, SEG_S, SEG_V);
    k_out0<<<gE32, TPB, 0, stream>>>(edge_idx, SEG_N, SEG_S, SEG_V, F_VS, Vv,
                                     F_WYS, F_WYV, F_OUT0, ARENA1);

    // h = silu(lora(xcat, m2_0)), xcat = [xm | out0] (K = 320)
    k_delta<<<gEW, TPB, 0, stream>>>(F_X, 256, 256, edge_mask, F_OUT0, 64, mod,
                                     m2_G0 + l * 256, m2_A0 + l * 1280,
                                     m2_B0 + l * 1024, 256, F_DELTA);
    gemm(ARENA1, 320, m2_W0 + l * 81920, 320, 256, F_DELTA, nullptr, nullptr,
         1.f, 1.f, 1, F_H, ARENA2, 256);

    // y = lora(h, m2_1);  x = (x + 0.5*u*y)/sqrt(1.25)
    k_delta<<<gEW, TPB, 0, stream>>>(F_H, 256, 256, nullptr, nullptr, 0, mod,
                                     m2_G1 + l * 256, m2_A1 + l * 1024,
                                     m2_B1 + l * 1024, 256, F_DELTA);
    gemm(ARENA2, 256, m2_W1 + l * 65536, 256, 256, F_DELTA, F_X, edge_cut,
         0.5f, invb, 0, F_X, BF_X, 256);

    if (l < 2) {
      k_vv<<<gE32, TPB, 0, stream>>>(F_WYS, F_WYV, F_VS, Vv,
                                     lin_V + l * 2048, Vv2);
      float* tmp = Vv; Vv = Vv2; Vv2 = tmp;
      hipMemsetAsync(F_VS, 0, (size_t)E_ * 32 * 4, stream);
    }
  }

  // ---- fused head: out = x @ (W_post @ W_final[:128]) ----
  gemm(BF_X, 256, F_WC, 256, 128, nullptr, nullptr, nullptr, 1.f, 1.f, 0,
       out, nullptr, 0);
}
